// TransformerDecoderOnly_62466004353312
// MI455X (gfx1250) — matrix operand order repaired and checked by reading, not compiled, not run on hardware
//
#include <hip/hip_runtime.h>
#include <hip/hip_bf16.h>
#include <math.h>

#define B_   2
#define S_   2048
#define V_   1024
#define D_   512
#define H_   8
#define L_   6
#define DFF_ 2048
#define DK_  64

typedef __attribute__((ext_vector_type(16))) _Float16 v16h;
typedef __attribute__((ext_vector_type(8)))  _Float16 v8h;
typedef __attribute__((ext_vector_type(8)))  float    v8f;
typedef int v4i __attribute__((vector_size(16)));

#define AS1 __attribute__((address_space(1)))
#define AS3 __attribute__((address_space(3)))

// CDNA5 async memory->LDS DMA path (ASYNCcnt tracked). Fallback: VGPR staging.
#if defined(__AMDGCN__) && __has_builtin(__builtin_amdgcn_global_load_async_to_lds_b128)
#define USE_ASYNC_LDS 1
#else
#define USE_ASYNC_LDS 0
#endif

__device__ __forceinline__ void wait_async_stage() {
#if USE_ASYNC_LDS
#if __has_builtin(__builtin_amdgcn_s_wait_asynccnt)
    __builtin_amdgcn_s_wait_asynccnt(0);
#else
    asm volatile("s_wait_asynccnt 0" ::: "memory");
#endif
#endif
}

// ---------------------------------------------------------------------------
// WMMA GEMM:  C[M,N] = A[M,K] * Bt[N,K]^T   (f16 in, f32 accumulate)
// Optional: +bias[n], +addC[m,n] (residual), relu, f16 or f32 output.
// Requires M%TBM==0, N%TBN==0, K%32==0 (true for every shape in this model).
// Double-buffered LDS; tiles staged with GLOBAL_LOAD_ASYNC_TO_LDS_B128.
// Wave tile = (TBM/WVM) x (TBN/WVN); threads = WVM*WVN*32.
// ---------------------------------------------------------------------------
#define BK 32
#define LDS_STR 40   // padded row stride (halfs): 80B rows -> conflict-free b128

template<int TBM, int TBN, int WVM, int WVN>
__global__ __launch_bounds__(WVM * WVN * 32)
void gemm_wmma(const _Float16* __restrict__ A, int lda,
               const _Float16* __restrict__ Bt, int ldb,
               const float* __restrict__ bias,
               const float* addC,
               float* Cf, _Float16* Ch,
               int ldc, int M, int N, int K, int relu)
{
    constexpr int WM  = TBM / WVM;     // wave tile rows   (32)
    constexpr int WN  = TBN / WVN;     // wave tile cols   (64)
    constexpr int AF  = WM / 16;       // A fragments per wave
    constexpr int BF  = WN / 16;       // B fragments per wave
    // staging: threads = 2*TBM, each thread copies 32B of A and 32B of B
    static_assert(WVM * WVN * 32 == 2 * TBM && TBN == TBM, "copy mapping");

    __shared__ _Float16 As[2][TBM * LDS_STR];
    __shared__ _Float16 Bs[2][TBN * LDS_STR];

    const int tileM = blockIdx.y * TBM;
    const int tileN = blockIdx.x * TBN;
    const int t    = threadIdx.x;
    const int lane = t & 31;
    const int wv   = t >> 5;
    const int lm   = lane & 15;
    const int g    = lane >> 4;
    const int wvM  = wv % WVM;
    const int wvN  = wv / WVM;

    const int crow = t >> 1;           // 0..TBM-1
    const int cseg = t & 1;            // 0..1 (16-half segment)
    const _Float16* gA = A  + (size_t)(tileM + crow) * lda + cseg * 16;
    const _Float16* gB = Bt + (size_t)(tileN + crow) * ldb + cseg * 16;
    const int soff = crow * LDS_STR + cseg * 16;

    auto stage = [&](int buf, int k0) {
        const _Float16* pa = gA + k0;
        const _Float16* pb = gB + k0;
        _Float16* sa = &As[buf][soff];
        _Float16* sb = &Bs[buf][soff];
#if USE_ASYNC_LDS
        __builtin_amdgcn_global_load_async_to_lds_b128(
            (AS1 v4i*)(v4i*)(void*)pa, (AS3 v4i*)(v4i*)sa, 0, 0);
        __builtin_amdgcn_global_load_async_to_lds_b128(
            (AS1 v4i*)(v4i*)(void*)pa, (AS3 v4i*)(v4i*)sa, 16, 0);
        __builtin_amdgcn_global_load_async_to_lds_b128(
            (AS1 v4i*)(v4i*)(void*)pb, (AS3 v4i*)(v4i*)sb, 0, 0);
        __builtin_amdgcn_global_load_async_to_lds_b128(
            (AS1 v4i*)(v4i*)(void*)pb, (AS3 v4i*)(v4i*)sb, 16, 0);
#else
        const uint4* va = (const uint4*)pa;
        const uint4* vb = (const uint4*)pb;
        uint4 a0 = va[0], a1 = va[1], b0 = vb[0], b1 = vb[1];
        ((uint4*)sa)[0] = a0; ((uint4*)sa)[1] = a1;
        ((uint4*)sb)[0] = b0; ((uint4*)sb)[1] = b1;
#endif
    };

    v8f acc[AF][BF];
#pragma unroll
    for (int i = 0; i < AF; ++i)
#pragma unroll
        for (int j = 0; j < BF; ++j)
#pragma unroll
            for (int r = 0; r < 8; ++r) acc[i][j][r] = 0.0f;

    stage(0, 0);
    wait_async_stage();
    __syncthreads();

    int buf = 0;
    for (int k0 = 0; k0 < K; k0 += BK) {
        if (k0 + BK < K) stage(buf ^ 1, k0 + BK);   // prefetch next tile (DMA)

        // ---- A fragments (ISA 16-bit A 16x32 layout): M=lm, K chunks by g
        v16h af[AF];
#pragma unroll
        for (int i = 0; i < AF; ++i) {
            const _Float16* ap = &As[buf][(wvM * WM + i * 16 + lm) * LDS_STR + g * 8];
            v8h lo = *(const v8h*)(ap);          // K = g*8 .. +7
            v8h hi = *(const v8h*)(ap + 16);     // K = 16+g*8 .. +7
            af[i] = __builtin_shufflevector(lo, hi,
                        0,1,2,3,4,5,6,7,8,9,10,11,12,13,14,15);
        }
#pragma unroll
        for (int j = 0; j < BF; ++j) {
            // ---- B fragment: N=lm, K = g*8 .. +7 and 16+g*8 .. +7 (same as A)
            const _Float16* bp = &Bs[buf][(wvN * WN + j * 16 + lm) * LDS_STR + g * 8];
            v8h lo = *(const v8h*)(bp);
            v8h hi = *(const v8h*)(bp + 16);
            v16h bf = __builtin_shufflevector(lo, hi,
                          0,1,2,3,4,5,6,7,8,9,10,11,12,13,14,15);
#pragma unroll
            for (int i = 0; i < AF; ++i)
                acc[i][j] = __builtin_amdgcn_wmma_f32_16x16x32_f16(
                                false, af[i], false, bf, (short)0, acc[i][j],
                                false, false);
        }
        wait_async_stage();
        __syncthreads();
        buf ^= 1;
    }

    // ---- store D (f32 C/D layout: VGPR r -> M = g*8+r, N = lm)
#pragma unroll
    for (int j = 0; j < BF; ++j) {
        const int n = tileN + wvN * WN + j * 16 + lm;
#pragma unroll
        for (int i = 0; i < AF; ++i) {
#pragma unroll
            for (int r = 0; r < 8; ++r) {
                const int m = tileM + wvM * WM + i * 16 + g * 8 + r;
                float val = acc[i][j][r];
                if (bias) val += bias[n];
                if (addC) val += addC[(size_t)m * ldc + n];
                if (relu) val = fmaxf(val, 0.0f);
                const size_t idx = (size_t)m * ldc + n;
                if (Ch) Ch[idx] = (_Float16)val;
                else    Cf[idx] = val;
            }
        }
    }
}

// ---------------------------------------------------------------------------
// Embedding lookup + sinusoidal positional encoding (f32 out)
// ---------------------------------------------------------------------------
__global__ void embed_kernel(const int* __restrict__ ids,
                             const float* __restrict__ emb,
                             float* __restrict__ x)
{
    int idx = blockIdx.x * blockDim.x + threadIdx.x;
    if (idx >= B_ * S_ * D_) return;
    int d   = idx & (D_ - 1);
    int row = idx >> 9;            // /D_
    int s   = row & (S_ - 1);
    int id  = ids[row];
    float freq = expf(-(float)(d & ~1) * (9.210340371976184f / (float)D_)); // ln(10000)/D
    float ang  = (float)s * freq;
    float pe   = (d & 1) ? cosf(ang) : sinf(ang);
    x[idx] = emb[(size_t)id * D_ + d] + pe;
}

// ---------------------------------------------------------------------------
// LayerNorm over D=512, writes f16 (GEMM A operand). One block per row.
// ---------------------------------------------------------------------------
__global__ __launch_bounds__(256)
void ln_kernel(const float* __restrict__ x, const float* __restrict__ w,
               const float* __restrict__ b, _Float16* __restrict__ out)
{
    __shared__ float red[256];
    const int r = blockIdx.x, t = threadIdx.x;
    const float* xr = x + (size_t)r * D_;
    float v0 = xr[t], v1 = xr[t + 256];

    red[t] = v0 + v1; __syncthreads();
    for (int s = 128; s > 0; s >>= 1) { if (t < s) red[t] += red[t + s]; __syncthreads(); }
    float mu = red[0] * (1.0f / D_);
    __syncthreads();

    float d0 = v0 - mu, d1 = v1 - mu;
    red[t] = d0 * d0 + d1 * d1; __syncthreads();
    for (int s = 128; s > 0; s >>= 1) { if (t < s) red[t] += red[t + s]; __syncthreads(); }
    float inv = rsqrtf(red[0] * (1.0f / D_) + 1e-5f);

    out[(size_t)r * D_ + t]       = (_Float16)(d0 * inv * w[t] + b[t]);
    out[(size_t)r * D_ + t + 256] = (_Float16)(d1 * inv * w[t + 256] + b[t + 256]);
}

// ---------------------------------------------------------------------------
// Masked softmax over one score row (causal + pad), scale fused; f16 probs out.
// ---------------------------------------------------------------------------
__global__ __launch_bounds__(256)
void softmax_kernel(const float* __restrict__ scores, const int* __restrict__ ids,
                    _Float16* __restrict__ p)
{
    __shared__ float red[256];
    const int r = blockIdx.x, t = threadIdx.x;
    const float scale = 0.125f;   // 1/sqrt(DK)

    float sv[8], ev[8];
    int   ok[8];
    float mx = -3.4e38f;
#pragma unroll
    for (int i = 0; i < 8; ++i) {
        int c = t + i * 256;
        ok[i] = (c <= r) && (ids[c] != 0);
        sv[i] = ok[i] ? scores[(size_t)r * S_ + c] * scale : -3.4e38f;
        mx = fmaxf(mx, sv[i]);
    }
    red[t] = mx; __syncthreads();
    for (int s = 128; s > 0; s >>= 1) { if (t < s) red[t] = fmaxf(red[t], red[t + s]); __syncthreads(); }
    float rowmax = red[0]; __syncthreads();

    float sum = 0.0f;
#pragma unroll
    for (int i = 0; i < 8; ++i) {
        ev[i] = ok[i] ? expf(sv[i] - rowmax) : 0.0f;
        sum += ev[i];
    }
    red[t] = sum; __syncthreads();
    for (int s = 128; s > 0; s >>= 1) { if (t < s) red[t] += red[t + s]; __syncthreads(); }
    float inv = red[0] > 0.0f ? 1.0f / red[0] : 0.0f;

#pragma unroll
    for (int i = 0; i < 8; ++i)
        p[(size_t)r * S_ + t + i * 256] = (_Float16)(ev[i] * inv);
}

// ---------------------------------------------------------------------------
// Weight convert: W f32 [K][N] row-major -> Wt f16 [N][K] (GEMM Bt format)
// ---------------------------------------------------------------------------
__global__ void transpose_w_kernel(const float* __restrict__ W,
                                   _Float16* __restrict__ Wt, int K, int N)
{
    int idx = blockIdx.x * blockDim.x + threadIdx.x;
    if (idx >= K * N) return;
    int k = idx / N, n = idx - k * N;
    Wt[(size_t)n * K + k] = (_Float16)W[idx];
}

// V head transpose: V16[b,k,h,d] -> Vt[d][k]  (Bt format for the P·V GEMM)
__global__ void vhead_transpose(const _Float16* __restrict__ v16,
                                _Float16* __restrict__ vt, int bOff, int h)
{
    int idx = blockIdx.x * blockDim.x + threadIdx.x;   // DK_*S_
    if (idx >= DK_ * S_) return;
    int d = idx / S_, kk = idx - d * S_;
    vt[idx] = v16[(size_t)(bOff + kk) * D_ + h * DK_ + d];
}

__global__ void cvt_f16_kernel(const float* __restrict__ in,
                               _Float16* __restrict__ out, int n)
{
    int i = blockIdx.x * blockDim.x + threadIdx.x;
    if (i < n) out[i] = (_Float16)in[i];
}

// ---------------------------------------------------------------------------
extern "C" void kernel_launch(void* const* d_in, const int* in_sizes, int n_in,
                              void* d_out, int out_size, void* d_ws, size_t ws_size,
                              hipStream_t stream)
{
    (void)in_sizes; (void)n_in; (void)out_size; (void)ws_size;

    const int*   ids  = (const int*)  d_in[0];
    const float* emb  = (const float*)d_in[1];
    const float* wq   = (const float*)d_in[2];
    const float* bq   = (const float*)d_in[3];
    const float* wk   = (const float*)d_in[4];
    const float* bk   = (const float*)d_in[5];
    const float* wv   = (const float*)d_in[6];
    const float* bv   = (const float*)d_in[7];
    const float* wo   = (const float*)d_in[8];
    const float* bo   = (const float*)d_in[9];
    const float* ln1w = (const float*)d_in[10];
    const float* ln1b = (const float*)d_in[11];
    const float* ln2w = (const float*)d_in[12];
    const float* ln2b = (const float*)d_in[13];
    const float* w1   = (const float*)d_in[14];
    const float* b1   = (const float*)d_in[15];
    const float* w2   = (const float*)d_in[16];
    const float* b2   = (const float*)d_in[17];
    const float* outw = (const float*)d_in[18];
    const float* outb = (const float*)d_in[19];
    float* logits = (float*)d_out;

    const int M = B_ * S_;   // 4096

    char* ws = (char*)d_ws;
    size_t off = 0;
    auto take = [&](size_t bytes) -> char* {
        char* p = ws + off;
        off += (bytes + 255) & ~(size_t)255;
        return p;
    };
    float*    xf32    = (float*)   take((size_t)M * D_   * 4);   //  8 MB
    _Float16* h16     = (_Float16*)take((size_t)M * D_   * 2);   //  4 MB
    _Float16* q16     = (_Float16*)take((size_t)M * D_   * 2);
    _Float16* k16     = (_Float16*)take((size_t)M * D_   * 2);
    _Float16* v16     = (_Float16*)take((size_t)M * D_   * 2);
    _Float16* o16     = (_Float16*)take((size_t)M * D_   * 2);
    _Float16* a16     = (_Float16*)take((size_t)M * DFF_ * 2);   // 16 MB
    _Float16* wt16    = (_Float16*)take((size_t)D_ * DFF_ * 2);  //  2 MB (max weight)
    float*    scoresf = (float*)   take((size_t)S_ * S_  * 4);   // 16 MB
    _Float16* p16     = (_Float16*)take((size_t)S_ * S_  * 2);   //  8 MB
    _Float16* vt16    = (_Float16*)take((size_t)DK_ * S_ * 2);   // 256 KB

    // big:   128x128 block tile, 8 waves (wave tile 32x64)
    // small: 64x64 block tile, 4 waves (wave tile 32x32) -- for N=64 (P*V)
    auto* gemm_big   = gemm_wmma<128, 128, 4, 2>;
    auto* gemm_small = gemm_wmma< 64,  64, 2, 2>;

    // 1) embedding + positional encoding
    embed_kernel<<<(M * D_ + 255) / 256, 256, 0, stream>>>(ids, emb, xf32);

    for (int l = 0; l < L_; ++l) {
        const float* Wq = wq + (size_t)l * D_ * D_;
        const float* Wk = wk + (size_t)l * D_ * D_;
        const float* Wv = wv + (size_t)l * D_ * D_;
        const float* Wo = wo + (size_t)l * D_ * D_;
        const float* W1 = w1 + (size_t)l * D_ * DFF_;
        const float* W2 = w2 + (size_t)l * DFF_ * D_;

        // h = LN1(x)
        ln_kernel<<<M, 256, 0, stream>>>(xf32, ln1w + l * D_, ln1b + l * D_, h16);

        // q/k/v = h @ W + b      (f16 out, [B,S,H,DK] == [M,D])
        transpose_w_kernel<<<(D_ * D_ + 255) / 256, 256, 0, stream>>>(Wq, wt16, D_, D_);
        gemm_big<<<dim3(D_ / 128, M / 128), 256, 0, stream>>>(
            h16, D_, wt16, D_, bq + l * D_, nullptr, nullptr, q16, D_, M, D_, D_, 0);
        transpose_w_kernel<<<(D_ * D_ + 255) / 256, 256, 0, stream>>>(Wk, wt16, D_, D_);
        gemm_big<<<dim3(D_ / 128, M / 128), 256, 0, stream>>>(
            h16, D_, wt16, D_, bk + l * D_, nullptr, nullptr, k16, D_, M, D_, D_, 0);
        transpose_w_kernel<<<(D_ * D_ + 255) / 256, 256, 0, stream>>>(Wv, wt16, D_, D_);
        gemm_big<<<dim3(D_ / 128, M / 128), 256, 0, stream>>>(
            h16, D_, wt16, D_, bv + l * D_, nullptr, nullptr, v16, D_, M, D_, D_, 0);

        // attention per (batch, head)
        for (int z = 0; z < B_ * H_; ++z) {
            const int b = z / H_, h = z % H_;
            const size_t hoff = (size_t)b * S_ * D_ + (size_t)h * DK_;
            // scores = Q @ K^T  (K head rows are already in Bt [N][K] format)
            gemm_big<<<dim3(S_ / 128, S_ / 128), 256, 0, stream>>>(
                q16 + hoff, D_, k16 + hoff, D_, nullptr, nullptr,
                scoresf, nullptr, S_, S_, S_, DK_, 0);
            // softmax with causal + pad mask, scale fused
            softmax_kernel<<<S_, 256, 0, stream>>>(scoresf, ids + b * S_, p16);
            // V head -> Bt format [DK][S]
            vhead_transpose<<<(DK_ * S_ + 255) / 256, 256, 0, stream>>>(v16, vt16, b * S_, h);
            // o = P @ V  (f16 out, scattered back into [B,S,H,DK])
            gemm_small<<<dim3(DK_ / 64, S_ / 64), 128, 0, stream>>>(
                p16, S_, vt16, S_, nullptr, nullptr, nullptr, o16 + hoff, D_, S_, DK_, S_, 0);
        }

        // x = o @ Wo + bo   (no residual -- faithful to reference)
        transpose_w_kernel<<<(D_ * D_ + 255) / 256, 256, 0, stream>>>(Wo, wt16, D_, D_);
        gemm_big<<<dim3(D_ / 128, M / 128), 256, 0, stream>>>(
            o16, D_, wt16, D_, bo + l * D_, nullptr, xf32, nullptr, D_, M, D_, D_, 0);

        // h2 = LN2(x)
        ln_kernel<<<M, 256, 0, stream>>>(xf32, ln2w + l * D_, ln2b + l * D_, h16);

        // ff = relu(h2 @ W1 + b1)   (f16 out)
        transpose_w_kernel<<<(D_ * DFF_ + 255) / 256, 256, 0, stream>>>(W1, wt16, D_, DFF_);
        gemm_big<<<dim3(DFF_ / 128, M / 128), 256, 0, stream>>>(
            h16, D_, wt16, D_, b1 + l * DFF_, nullptr, nullptr, a16, DFF_, M, DFF_, D_, 1);

        // x = x + ff @ W2 + b2     (residual fused into GEMM epilogue)
        transpose_w_kernel<<<(DFF_ * D_ + 255) / 256, 256, 0, stream>>>(W2, wt16, DFF_, D_);
        gemm_big<<<dim3(D_ / 128, M / 128), 256, 0, stream>>>(
            a16, DFF_, wt16, DFF_, b2 + l * D_, xf32, xf32, nullptr, D_, M, D_, DFF_, 0);
    }

    // logits = x @ out_w + out_b
    cvt_f16_kernel<<<(M * D_ + 255) / 256, 256, 0, stream>>>(xf32, h16, M * D_);
    transpose_w_kernel<<<(D_ * V_ + 255) / 256, 256, 0, stream>>>(outw, wt16, D_, V_);
    gemm_big<<<dim3(V_ / 128, M / 128), 256, 0, stream>>>(
        h16, D_, wt16, D_, outb, nullptr, logits, nullptr, V_, M, V_, D_, 0);
}
